// TGNMemory_22428319220240
// MI455X (gfx1250) — matrix operand order repaired and checked by reading, not compiled, not run on hardware
//
#include <hip/hip_runtime.h>
#include <hip/hip_bf16.h>
#include <math.h>

#define Nn 131072
#define Ee 262144
#define Dd 128
#define Tt 32
#define KK 416            // 2D + R + T

typedef __attribute__((ext_vector_type(16))) __bf16 v16bf;
typedef __attribute__((ext_vector_type(8)))  float  v8f;

union ABf { uint4 u[2]; v16bf v; };

__device__ __forceinline__ unsigned short f2bf(float f) {
    union { float f; unsigned u; } x; x.f = f;
    unsigned r = x.u + 0x7FFFu + ((x.u >> 16) & 1u);   // round-to-nearest-even
    return (unsigned short)(r >> 16);
}

__device__ __forceinline__ float sigm(float x) {
    return 1.0f / (1.0f + __expf(-x));
}

// ---------------------------------------------------------------------------
// Pack a row-major weight matrix W[rows, K] into bf16 WMMA B-fragments.
// Fragment (kt, nt): 32 lanes x 16 bf16, lane l holds column nt*16+(l&15),
// K = kt*32 + (l>>4)*8 + (i&7) + (i>>3)*16  (matches 16-bit B 32x16 layout, ISA 7.12.2).
// ---------------------------------------------------------------------------
__global__ void pack_w(const float* __restrict__ W, unsigned short* __restrict__ out,
                       int ktiles, int ntiles, int K) {
    int gid = blockIdx.x * blockDim.x + threadIdx.x;
    int total = ktiles * ntiles * 512;
    if (gid >= total) return;
    int f    = gid >> 9;
    int r    = gid & 511;
    int lane = r >> 4;
    int i    = r & 15;
    int kt   = f / ntiles;
    int nt   = f - kt * ntiles;
    int j    = nt * 16 + (lane & 15);
    int k    = kt * 32 + (lane >> 4) * 8 + (i & 7) + (i >> 3) * 16;
    out[gid] = f2bf(W[(size_t)j * K + k]);
}

// ---------------------------------------------------------------------------
// Message stage: 64 events per block, 4 waves, 16 events per wave.
// LDS row per event (bf16): [mem_s 0..127][mem_d 128..255][raw 256..383]
//                           [tenc_s 384..415][tenc_d 416..447], stride 472.
// ---------------------------------------------------------------------------
#define XSTR 472

__global__ __launch_bounds__(128) void tgn_msg_kernel(
    const float* __restrict__ memory, const float* __restrict__ raw_msg,
    const float* __restrict__ time_w, const float* __restrict__ time_b,
    const unsigned short* __restrict__ wp_s, const unsigned short* __restrict__ wp_d,
    const float* __restrict__ b_msg_s, const float* __restrict__ b_msg_d,
    const int* __restrict__ src, const int* __restrict__ dst,
    const int* __restrict__ tstamp, const int* __restrict__ last_update,
    float* __restrict__ sums, float* __restrict__ cnt, int* __restrict__ tmax)
{
    __shared__ unsigned short sX[64 * XSTR];
    __shared__ int   sNode[2][64];
    __shared__ float sTr[2][64];

    const int tid = threadIdx.x;
    const int eb  = blockIdx.x * 64;

    if (tid < 64) {
        int e  = eb + tid;
        int s  = src[e];
        int d2 = dst[e];
        int tv = tstamp[e];
        sNode[0][tid] = s;
        sNode[1][tid] = d2;
        float tf = (float)tv;
        sTr[0][tid] = tf - (float)last_update[s];
        sTr[1][tid] = tf - (float)last_update[d2];
        (void)__hip_atomic_fetch_add(&cnt[s], 1.0f, __ATOMIC_RELAXED, __HIP_MEMORY_SCOPE_AGENT);
        (void)__hip_atomic_fetch_add(&cnt[d2], 1.0f, __ATOMIC_RELAXED, __HIP_MEMORY_SCOPE_AGENT);
        atomicMax(&tmax[s], tv);
        atomicMax(&tmax[d2], tv);
    }
    __syncthreads();

    // time encodings: cos(t_rel * w + b) for both sides
    for (int idx = tid; idx < 64 * 64; idx += 128) {
        int el = idx >> 6, j = idx & 63, enc = j >> 5, jj = j & 31;
        float v = __cosf(sTr[enc][el] * time_w[jj] + time_b[jj]);
        sX[el * XSTR + 384 + enc * 32 + jj] = f2bf(v);
    }

    // gather + convert mem_s / mem_d / raw rows into LDS (float4 granules)
    for (int idx = tid; idx < 64 * 96; idx += 128) {
        int el = idx / 96, p = idx - el * 96;
        int region = p >> 5, c4 = p & 31;
        const float* sp;
        if (region == 0)      sp = memory  + (size_t)sNode[0][el] * Dd + c4 * 4;
        else if (region == 1) sp = memory  + (size_t)sNode[1][el] * Dd + c4 * 4;
        else                  sp = raw_msg + (size_t)(eb + el)    * Dd + c4 * 4;
        float4 v = *(const float4*)sp;
        uint2 pk;
        pk.x = (unsigned)f2bf(v.x) | ((unsigned)f2bf(v.y) << 16);
        pk.y = (unsigned)f2bf(v.z) | ((unsigned)f2bf(v.w) << 16);
        *(uint2*)&sX[el * XSTR + region * 128 + c4 * 4] = pk;
    }
    __syncthreads();

    const int wave  = tid >> 5;
    const int lane  = tid & 31;
    const int lhalf = lane >> 4;
    const int lcol  = lane & 15;
    const int row0  = wave * 16;
    const unsigned short* aRow = &sX[(row0 + lcol) * XSTR];

    for (int m = 0; m < 2; ++m) {
        const unsigned short* wp   = m ? wp_d : wp_s;
        const float*          bias = m ? b_msg_d : b_msg_s;

        v8f acc[8];
#pragma unroll
        for (int nt = 0; nt < 8; ++nt)
            acc[nt] = (v8f){0.f,0.f,0.f,0.f,0.f,0.f,0.f,0.f};

        for (int kt = 0; kt < 13; ++kt) {
            // K-tile base offset in the LDS row (msg_d swaps mem blocks, uses tenc_d)
            int kb;
            if (m == 0) kb = kt * 32;
            else        kb = (kt < 4)  ? 128 + kt * 32
                           : (kt < 8)  ? (kt - 4) * 32
                           : (kt < 12) ? kt * 32
                                       : 416;
            ABf a;
            const uint4* ap = (const uint4*)(aRow + kb + lhalf * 8);
            a.u[0] = ap[0];   // K + 0..7
            a.u[1] = ap[2];   // K + 16..23
#pragma unroll
            for (int nt = 0; nt < 8; ++nt) {
                ABf b;
                const uint4* bp = ((const uint4*)wp) + (size_t)(kt * 8 + nt) * 64 + lane * 2;
                b.u[0] = bp[0];
                b.u[1] = bp[1];
                acc[nt] = __builtin_amdgcn_wmma_f32_16x16x32_bf16(
                    false, a.v, false, b.v, (short)0, acc[nt], false, false);
            }
        }

        int nodes[8];
#pragma unroll
        for (int r = 0; r < 8; ++r)
            nodes[r] = sNode[m][row0 + r + lhalf * 8];

#pragma unroll
        for (int nt = 0; nt < 8; ++nt) {
            int col  = nt * 16 + lcol;
            float bv = bias[col];
#pragma unroll
            for (int r = 0; r < 8; ++r) {
                float v = acc[nt][r] + bv;
                v = v > 0.0f ? v : 0.0f;   // ReLU
                (void)__hip_atomic_fetch_add(&sums[(size_t)nodes[r] * Dd + col], v,
                                             __ATOMIC_RELAXED, __HIP_MEMORY_SCOPE_AGENT);
            }
        }
    }
}

// ---------------------------------------------------------------------------
// GRU stage: 64 nodes per block, 4 waves, 16 nodes per wave.
// ---------------------------------------------------------------------------
#define ASTR 136   // 128 + 8 pad -> 16 distinct LDS banks across M-lanes

__global__ __launch_bounds__(128) void tgn_gru_kernel(
    const float* __restrict__ memory, const float* __restrict__ sums,
    const float* __restrict__ cnt, const int* __restrict__ tmax,
    const unsigned short* __restrict__ wp_ih, const unsigned short* __restrict__ wp_hh,
    const float* __restrict__ b_ih, const float* __restrict__ b_hh,
    float* __restrict__ out_mem, float* __restrict__ out_lu)
{
    __shared__ unsigned short sA[64 * ASTR];   // aggr rows (bf16)
    __shared__ unsigned short sM[64 * ASTR];   // memory rows (bf16)

    const int tid = threadIdx.x;
    const int nb  = blockIdx.x * 64;

    for (int idx = tid; idx < 64 * 32; idx += 128) {
        int nl = idx >> 5, c4 = (idx & 31) * 4;
        int node = nb + nl;
        float c   = cnt[node];
        float inv = 1.0f / (c > 1.0f ? c : 1.0f);
        float4 s4 = *(const float4*)(sums   + (size_t)node * Dd + c4);
        float4 m4 = *(const float4*)(memory + (size_t)node * Dd + c4);
        uint2 pa, pm;
        pa.x = (unsigned)f2bf(s4.x * inv) | ((unsigned)f2bf(s4.y * inv) << 16);
        pa.y = (unsigned)f2bf(s4.z * inv) | ((unsigned)f2bf(s4.w * inv) << 16);
        pm.x = (unsigned)f2bf(m4.x)       | ((unsigned)f2bf(m4.y)       << 16);
        pm.y = (unsigned)f2bf(m4.z)       | ((unsigned)f2bf(m4.w)       << 16);
        *(uint2*)&sA[nl * ASTR + c4] = pa;
        *(uint2*)&sM[nl * ASTR + c4] = pm;
    }
    if (tid < 64) {
        int node = nb + tid;
        float c = cnt[node];
        out_lu[node] = (c > 0.0f) ? (float)tmax[node] : 0.0f;
    }
    __syncthreads();

    const int wave  = tid >> 5;
    const int lane  = tid & 31;
    const int lhalf = lane >> 4;
    const int lcol  = lane & 15;
    const int row0  = wave * 16;
    const unsigned short* aRow = &sA[(row0 + lcol) * ASTR];
    const unsigned short* mRow = &sM[(row0 + lcol) * ASTR];

    int nodes[8];
#pragma unroll
    for (int r = 0; r < 8; ++r)
        nodes[r] = nb + row0 + r + lhalf * 8;

    for (int nt = 0; nt < 8; ++nt) {
        v8f gi[3], gh[3];
#pragma unroll
        for (int g = 0; g < 3; ++g) {
            gi[g] = (v8f){0.f,0.f,0.f,0.f,0.f,0.f,0.f,0.f};
            gh[g] = (v8f){0.f,0.f,0.f,0.f,0.f,0.f,0.f,0.f};
        }
#pragma unroll
        for (int kt = 0; kt < 4; ++kt) {
            ABf a, h;
            const uint4* ap = (const uint4*)(aRow + kt * 32 + lhalf * 8);
            a.u[0] = ap[0]; a.u[1] = ap[2];
            const uint4* hp = (const uint4*)(mRow + kt * 32 + lhalf * 8);
            h.u[0] = hp[0]; h.u[1] = hp[2];
#pragma unroll
            for (int g = 0; g < 3; ++g) {
                int ct = nt + g * 8;   // r: cols [0,128), z: [128,256), n: [256,384)
                ABf Bi, Bh;
                const uint4* bi = ((const uint4*)wp_ih) + (size_t)(kt * 24 + ct) * 64 + lane * 2;
                const uint4* bh = ((const uint4*)wp_hh) + (size_t)(kt * 24 + ct) * 64 + lane * 2;
                Bi.u[0] = bi[0]; Bi.u[1] = bi[1];
                Bh.u[0] = bh[0]; Bh.u[1] = bh[1];
                gi[g] = __builtin_amdgcn_wmma_f32_16x16x32_bf16(
                    false, a.v, false, Bi.v, (short)0, gi[g], false, false);
                gh[g] = __builtin_amdgcn_wmma_f32_16x16x32_bf16(
                    false, h.v, false, Bh.v, (short)0, gh[g], false, false);
            }
        }

        int col = nt * 16 + lcol;
        float bir = b_ih[col], biz = b_ih[col + 128], bin = b_ih[col + 256];
        float bhr = b_hh[col], bhz = b_hh[col + 128], bhn = b_hh[col + 256];
#pragma unroll
        for (int r = 0; r < 8; ++r) {
            float rr = sigm(gi[0][r] + bir + gh[0][r] + bhr);
            float zz = sigm(gi[1][r] + biz + gh[1][r] + bhz);
            float nn = tanhf(gi[2][r] + bin + rr * (gh[2][r] + bhn));
            float mv = memory[(size_t)nodes[r] * Dd + col];
            out_mem[(size_t)nodes[r] * Dd + col] = (1.0f - zz) * nn + zz * mv;
        }
    }
}

// ---------------------------------------------------------------------------
extern "C" void kernel_launch(void* const* d_in, const int* in_sizes, int n_in,
                              void* d_out, int out_size, void* d_ws, size_t ws_size,
                              hipStream_t stream) {
    (void)in_sizes; (void)n_in; (void)out_size; (void)ws_size;

    const float* memory      = (const float*)d_in[0];
    const float* raw_msg     = (const float*)d_in[1];
    const float* time_w      = (const float*)d_in[2];
    const float* time_b      = (const float*)d_in[3];
    const float* W_msg_s     = (const float*)d_in[4];
    const float* b_msg_s     = (const float*)d_in[5];
    const float* W_msg_d     = (const float*)d_in[6];
    const float* b_msg_d     = (const float*)d_in[7];
    const float* W_ih        = (const float*)d_in[8];
    const float* W_hh        = (const float*)d_in[9];
    const float* b_ih        = (const float*)d_in[10];
    const float* b_hh        = (const float*)d_in[11];
    const int*   src         = (const int*)d_in[12];
    const int*   dst         = (const int*)d_in[13];
    const int*   tstamp      = (const int*)d_in[14];
    const int*   last_update = (const int*)d_in[15];

    // workspace layout (bytes)
    char* ws = (char*)d_ws;
    const size_t sz_sums = (size_t)Nn * Dd * sizeof(float);     // 64 MB
    const size_t sz_cnt  = (size_t)Nn * sizeof(float);
    const size_t sz_tmax = (size_t)Nn * sizeof(int);
    const size_t sz_wmsg = (size_t)13 * 8 * 512 * sizeof(unsigned short);  // 53248 bf16
    const size_t sz_wgru = (size_t)4 * 24 * 512 * sizeof(unsigned short);  // 49152 bf16

    float*          sums  = (float*)(ws);
    float*          cnt   = (float*)(ws + sz_sums);
    int*            tmax  = (int*)  (ws + sz_sums + sz_cnt);
    unsigned short* wp_s  = (unsigned short*)(ws + sz_sums + sz_cnt + sz_tmax);
    unsigned short* wp_d  = (unsigned short*)((char*)wp_s  + sz_wmsg);
    unsigned short* wp_ih = (unsigned short*)((char*)wp_d  + sz_wmsg);
    unsigned short* wp_hh = (unsigned short*)((char*)wp_ih + sz_wgru);

    // zero accumulators (sums + cnt + tmax are contiguous)
    hipMemsetAsync(ws, 0, sz_sums + sz_cnt + sz_tmax, stream);

    // pack weights to bf16 WMMA fragments
    pack_w<<<208, 256, 0, stream>>>(W_msg_s, wp_s, 13, 8, KK);
    pack_w<<<208, 256, 0, stream>>>(W_msg_d, wp_d, 13, 8, KK);
    pack_w<<<192, 256, 0, stream>>>(W_ih, wp_ih, 4, 24, Dd);
    pack_w<<<192, 256, 0, stream>>>(W_hh, wp_hh, 4, 24, Dd);

    // message + scatter stage: E/64 blocks
    tgn_msg_kernel<<<Ee / 64, 128, 0, stream>>>(
        memory, raw_msg, time_w, time_b, wp_s, wp_d, b_msg_s, b_msg_d,
        src, dst, tstamp, last_update, sums, cnt, tmax);

    // GRU stage: N/64 blocks
    float* out_mem = (float*)d_out;
    float* out_lu  = out_mem + (size_t)Nn * Dd;
    tgn_gru_kernel<<<Nn / 64, 128, 0, stream>>>(
        memory, sums, cnt, tmax, wp_ih, wp_hh, b_ih, b_hh, out_mem, out_lu);
}
